// ResidualVQEMA_11055245820133
// MI455X (gfx1250) — matrix operand order repaired and checked by reading, not compiled, not run on hardware
//
#include <hip/hip_runtime.h>

#define NB 4
#define NE 1024
#define DD 128
#define TT 2048
#define LDST 136   // halves per padded LDS row (128 + 8 pad -> conflict-free b128)

typedef _Float16 v16h __attribute__((ext_vector_type(16)));
typedef _Float16 v8h  __attribute__((ext_vector_type(8)));
typedef float    v8f  __attribute__((ext_vector_type(8)));

// async-to-LDS builtin operand types: 16-byte int vector in AS1 (global) / AS3 (LDS)
typedef int i4v __attribute__((vector_size(16)));
typedef __attribute__((address_space(1))) i4v gi4;
typedef __attribute__((address_space(3))) i4v si4;

// ---------------- prelude: books f32 -> f16 (workspace) + bias = 0.5*||e||^2 ----
__global__ __launch_bounds__(256) void vq_prep(const float* __restrict__ books,
                                               _Float16* __restrict__ books_h,
                                               float* __restrict__ bias) {
  int gw   = (blockIdx.x * 256 + threadIdx.x) >> 5;  // one wave per codebook row, 4096 rows
  int lane = threadIdx.x & 31;
  const float* rp = books + (size_t)gw * DD;
  _Float16*    hp = books_h + (size_t)gw * DD;
  float s = 0.f;
#pragma unroll
  for (int j = 0; j < 4; ++j) {
    float v = rp[lane + 32 * j];
    hp[lane + 32 * j] = (_Float16)v;
    s += v * v;
  }
#pragma unroll
  for (int off = 16; off >= 1; off >>= 1) s += __shfl_xor(s, off, 32);
  if (lane == 0) bias[gw] = 0.5f * s;
}

// ---------------- main fused RVQ kernel ---------------------------------------
// grid: 256 blocks (32 b * 8 t-tiles of 256), 256 threads = 8 waves, 32 tokens/wave
// (two 16-row A tiles per wave sharing B fragments -> 2-way WMMA ILP, half the
//  LDS B traffic per FLOP; B fragments double-buffered across n-chunks).
__global__ __launch_bounds__(256) void vq_main(const float* __restrict__ z,
                                               const float* __restrict__ books,
                                               const _Float16* __restrict__ books_h,
                                               const float* __restrict__ biasg,
                                               float* __restrict__ out) {
  extern __shared__ _Float16 smem[];
  _Float16* lds_book = smem;                        // NE * LDST halves (272 KB)
  float*    lds_bias = (float*)(smem + NE * LDST);  // NE floats (4 KB)

  const int tid  = threadIdx.x;
  const int lane = tid & 31;
  const int wave = tid >> 5;
  const int g    = lane >> 4;   // half-wave group (selects dim sub-block per WMMA layout)
  const int myr  = lane & 15;   // this lane's token row within a 16-row tile

  const int b    = blockIdx.x >> 3;
  const int t0   = (blockIdx.x & 7) * 256;
  const int tA   = t0 + wave * 32 + myr;   // tile0 token; tile1 token = tA + 16

  // residual / q_sum in registers, pre-swizzled for the 16-bit A layout:
  // res[c*8+i] = x[row=myr][dim = c*16 + g*8 + i]
  const float* zb = z + (size_t)b * DD * TT + tA;
  float res0[64], qs0[64], res1[64], qs1[64];
#pragma unroll
  for (int c = 0; c < 8; ++c)
#pragma unroll
    for (int i = 0; i < 8; ++i) {
      int d = c * 16 + g * 8 + i;
      res0[c * 8 + i] = zb[(size_t)d * TT];
      res1[c * 8 + i] = zb[(size_t)d * TT + 16];
      qs0[c * 8 + i] = 0.f;
      qs1[c * 8 + i] = 0.f;
    }

  for (int book = 0; book < NB; ++book) {
    __syncthreads();  // previous round's LDS reads done
    // ---- stage f16 codebook into padded LDS (async, ASYNCcnt-tracked) ----
    const _Float16* bh = books_h + (size_t)book * (NE * DD);
#if __has_builtin(__builtin_amdgcn_global_load_async_to_lds_b128)
    for (int it = tid; it < (NE * DD / 8); it += 256) {
      int row = it >> 4;
      int ch  = it & 15;
      const _Float16* gp = bh + it * 8;
      _Float16*       lp = lds_book + row * LDST + ch * 8;
      __builtin_amdgcn_global_load_async_to_lds_b128(
          (gi4*)(size_t)gp,
          (si4*)(unsigned)(size_t)(void*)lp, 0, 0);
    }
#if __has_builtin(__builtin_amdgcn_s_wait_asynccnt)
    __builtin_amdgcn_s_wait_asynccnt(0);
#else
    asm volatile("s_wait_asynccnt 0" ::: "memory");
#endif
#else
    for (int it = tid; it < (NE * DD / 8); it += 256) {
      int row = it >> 4;
      int ch  = it & 15;
      *(v8h*)(lds_book + row * LDST + ch * 8) = *(const v8h*)(bh + it * 8);
    }
#endif
    for (int it = tid; it < NE; it += 256)
      lds_bias[it] = biasg[book * NE + it];
    __syncthreads();

    // ---- A fragments (16x32 f16 per K-step) from the register residuals ----
    v16h aL[4], aH[4];
#pragma unroll
    for (int kk = 0; kk < 4; ++kk)
#pragma unroll
      for (int h = 0; h < 16; ++h) {
        aL[kk][h] = (_Float16)res0[kk * 16 + h];
        aH[kk][h] = (_Float16)res1[kk * 16 + h];
      }

    float best0[8], best1[8];
    int   bidx0[8], bidx1[8];
#pragma unroll
    for (int r = 0; r < 8; ++r) {
      best0[r] = -3.402823466e38f; bidx0[r] = 0;
      best1[r] = -3.402823466e38f; bidx1[r] = 0;
    }

    const _Float16* bpBase = lds_book + g * 8;

    // load the 4 B fragments (K-steps) of column-chunk nn into bf[]
    auto loadFrags = [&](int nn, v16h* bf) {
      const _Float16* bp = bpBase + (size_t)(nn * 16 + myr) * LDST;
#pragma unroll
      for (int kk = 0; kk < 4; ++kk) {
        v8h lo = *(const v8h*)(bp + kk * 32);       // K = kk*32 + g*8 + 0..7
        v8h hi = *(const v8h*)(bp + kk * 32 + 16);  // K = kk*32 + 16 + g*8 + 0..7
        bf[kk] = __builtin_shufflevector(lo, hi, 0,1,2,3,4,5,6,7,8,9,10,11,12,13,14,15);
      }
    };
    // two independent WMMA chains sharing bf[kk], then fused bias+argmax update
    auto computeChunk = [&](int nn, const v16h* bf) {
      int col = nn * 16 + myr;
      float bv = lds_bias[col];
      v8f acc0 = {}, acc1 = {};
#pragma unroll
      for (int kk = 0; kk < 4; ++kk) {
        acc0 = __builtin_amdgcn_wmma_f32_16x16x32_f16(false, aL[kk], false, bf[kk],
                                                      (short)0, acc0, false, false);
        acc1 = __builtin_amdgcn_wmma_f32_16x16x32_f16(false, aH[kk], false, bf[kk],
                                                      (short)0, acc1, false, false);
      }
#pragma unroll
      for (int r = 0; r < 8; ++r) {     // lane holds rows 8g+r, column `col`
        float s0 = acc0[r] - bv;
        bool c0 = (s0 > best0[r]);
        best0[r] = c0 ? s0 : best0[r];
        bidx0[r] = c0 ? col : bidx0[r];
        float s1 = acc1[r] - bv;
        bool c1 = (s1 > best1[r]);
        best1[r] = c1 ? s1 : best1[r];
        bidx1[r] = c1 ? col : bidx1[r];
      }
    };

    // double-buffered n-chunk loop: prefetch chunk n+1 while computing chunk n
    v16h fA[4], fB[4];
    loadFrags(0, fA);
    for (int nn = 0; nn < 64; nn += 2) {
      loadFrags(nn + 1, fB);
      computeChunk(nn, fA);
      loadFrags((nn + 2) & 63, fA);   // wraps on last iter (result unused, in-bounds)
      computeChunk(nn + 1, fB);
    }

    // ---- reduce (val,idx) across the 16 lanes of each half-wave ----
#pragma unroll
    for (int off = 8; off >= 1; off >>= 1) {
#pragma unroll
      for (int r = 0; r < 8; ++r) {
        float ov0 = __shfl_xor(best0[r], off, 32);
        int   oi0 = __shfl_xor(bidx0[r], off, 32);
        bool tk0 = (ov0 > best0[r]) || (ov0 == best0[r] && oi0 < bidx0[r]);
        best0[r] = tk0 ? ov0 : best0[r];
        bidx0[r] = tk0 ? oi0 : bidx0[r];
        float ov1 = __shfl_xor(best1[r], off, 32);
        int   oi1 = __shfl_xor(bidx1[r], off, 32);
        bool tk1 = (ov1 > best1[r]) || (ov1 == best1[r] && oi1 < bidx1[r]);
        best1[r] = tk1 ? ov1 : best1[r];
        bidx1[r] = tk1 ? oi1 : bidx1[r];
      }
    }

    // ---- broadcast winner index for this lane's rows ----
    int srcl = (myr & 8) ? 16 : 0;   // rows 0-7 live in lanes 0-15, rows 8-15 in 16-31
    int rsub = myr & 7;
    int idx0 = 0, idx1 = 0;
#pragma unroll
    for (int r = 0; r < 8; ++r) {
      int v0 = __shfl(bidx0[r], srcl, 32);
      int v1 = __shfl(bidx1[r], srcl, 32);
      idx0 = (rsub == r) ? v0 : idx0;
      idx1 = (rsub == r) ? v1 : idx1;
    }

    // ---- gather exact f32 codebook rows; update residuals and q_sums ----
    const float* qp0 = books + ((size_t)book * NE + idx0) * DD + g * 8;
    const float* qp1 = books + ((size_t)book * NE + idx1) * DD + g * 8;
#pragma unroll
    for (int c = 0; c < 8; ++c) {
      float4 a0 = *(const float4*)(qp0 + c * 16);
      float4 a1 = *(const float4*)(qp0 + c * 16 + 4);
      float4 b0 = *(const float4*)(qp1 + c * 16);
      float4 b1 = *(const float4*)(qp1 + c * 16 + 4);
      res0[c*8+0] -= a0.x; qs0[c*8+0] += a0.x;
      res0[c*8+1] -= a0.y; qs0[c*8+1] += a0.y;
      res0[c*8+2] -= a0.z; qs0[c*8+2] += a0.z;
      res0[c*8+3] -= a0.w; qs0[c*8+3] += a0.w;
      res0[c*8+4] -= a1.x; qs0[c*8+4] += a1.x;
      res0[c*8+5] -= a1.y; qs0[c*8+5] += a1.y;
      res0[c*8+6] -= a1.z; qs0[c*8+6] += a1.z;
      res0[c*8+7] -= a1.w; qs0[c*8+7] += a1.w;
      res1[c*8+0] -= b0.x; qs1[c*8+0] += b0.x;
      res1[c*8+1] -= b0.y; qs1[c*8+1] += b0.y;
      res1[c*8+2] -= b0.z; qs1[c*8+2] += b0.z;
      res1[c*8+3] -= b0.w; qs1[c*8+3] += b0.w;
      res1[c*8+4] -= b1.x; qs1[c*8+4] += b1.x;
      res1[c*8+5] -= b1.y; qs1[c*8+5] += b1.y;
      res1[c*8+6] -= b1.z; qs1[c*8+6] += b1.z;
      res1[c*8+7] -= b1.w; qs1[c*8+7] += b1.w;
    }
  }

  // ---- write q_sum back in [B, D, T] layout ----
  float* ob = out + (size_t)b * DD * TT + tA;
#pragma unroll
  for (int c = 0; c < 8; ++c)
#pragma unroll
    for (int i = 0; i < 8; ++i) {
      int d = c * 16 + g * 8 + i;
      ob[(size_t)d * TT]      = qs0[c * 8 + i];
      ob[(size_t)d * TT + 16] = qs1[c * 8 + i];
    }
}

extern "C" void kernel_launch(void* const* d_in, const int* in_sizes, int n_in,
                              void* d_out, int out_size, void* d_ws, size_t ws_size,
                              hipStream_t stream) {
  const float* z     = (const float*)d_in[0];   // [32,128,2048] f32
  const float* books = (const float*)d_in[1];   // [4,1024,128] f32
  float* out = (float*)d_out;

  _Float16* books_h = (_Float16*)d_ws;                                  // 1 MB
  float*    bias    = (float*)((char*)d_ws + (size_t)NB * NE * DD * 2); // 16 KB

  vq_prep<<<512, 256, 0, stream>>>(books, books_h, bias);

  size_t shmem = (size_t)NE * LDST * 2 + (size_t)NE * 4;  // 282,624 B (<320 KB WGP LDS)
  (void)hipFuncSetAttribute(reinterpret_cast<const void*>(vq_main),
                            hipFuncAttributeMaxDynamicSharedMemorySize, (int)shmem);
  vq_main<<<256, 256, shmem, stream>>>(z, books, books_h, bias, out);
}
